// MultiHeadAttention_23837068493123
// MI455X (gfx1250) — matrix operand order repaired and checked by reading, not compiled, not run on hardware
//
#include <hip/hip_runtime.h>

typedef _Float16 f16_t;
typedef __attribute__((ext_vector_type(16))) _Float16 v16h;
typedef __attribute__((ext_vector_type(8)))  _Float16 v8h;
typedef __attribute__((ext_vector_type(8)))  float    v8f;

#define BDIM 2
#define NSEQ 2048
#define DMODEL 1024
#define NHEAD 16
#define DK 64

// ---- CDNA5 async global->LDS copy (ASYNCcnt path), with safe fallback -----
#if defined(__HIP_DEVICE_COMPILE__) && defined(__has_builtin)
#if __has_builtin(__builtin_amdgcn_global_load_async_to_lds_b128)
#define HAVE_ASYNC_CP 1
#endif
#if __has_builtin(__builtin_amdgcn_s_wait_asynccnt)
#define HAVE_WAIT_ASYNC_BUILTIN 1
#endif
#endif
#ifndef HAVE_ASYNC_CP
#define HAVE_ASYNC_CP 0
#endif
#ifndef HAVE_WAIT_ASYNC_BUILTIN
#define HAVE_WAIT_ASYNC_BUILTIN 0
#endif

// The builtin expects int4-typed pointers:
//   (int4 addrspace(1)* src, int4 addrspace(3)* dst, imm offset, imm cpol)
typedef __attribute__((vector_size(4 * sizeof(int)))) int i32x4;
typedef __attribute__((address_space(1))) i32x4 gi32x4_t;
typedef __attribute__((address_space(3))) i32x4 li32x4_t;

__device__ __forceinline__ void cp16_async(f16_t* lds, const f16_t* g) {
#if HAVE_ASYNC_CP
  __builtin_amdgcn_global_load_async_to_lds_b128(
      (gi32x4_t*)(g), (li32x4_t*)(lds), /*offset=*/0, /*cpol=*/0);
#else
  *(v8h*)lds = *(const v8h*)g;
#endif
}

__device__ __forceinline__ void async_wait0() {
#if HAVE_ASYNC_CP
#if HAVE_WAIT_ASYNC_BUILTIN
  __builtin_amdgcn_s_wait_asynccnt(0);
#else
  asm volatile("s_wait_asynccnt 0" ::: "memory");
#endif
#endif
}

__device__ __forceinline__ v8f wmma_f16(v16h a, v16h b, v8f c) {
  return __builtin_amdgcn_wmma_f32_16x16x32_f16(
      /*neg_a=*/false, a, /*neg_b=*/false, b,
      /*c_mod=*/(short)0, c, /*reuse_a=*/false, /*reuse_b=*/false);
}

// A fragment: 16x32 (MxK). tile base is row-major [m][k], ld in halves.
// Lane m = lane&15; element e<8 -> K = 8*hi + e ; e>=8 -> K = 16 + 8*hi + (e-8)
__device__ __forceinline__ v16h ld_frag_a(const f16_t* base, int ld, int kbase) {
  int lane = threadIdx.x & 31;
  int m = lane & 15, hi = lane >> 4;
  const f16_t* p = base + m * ld + kbase;
  union { v16h v; v8h h[2]; } u;
  u.h[0] = *(const v8h*)(p + hi * 8);
  u.h[1] = *(const v8h*)(p + 16 + hi * 8);
  return u.v;
}

// B fragment: 32x16 (KxN). tile stored transposed [n][k], ld in halves.
// Lane n = lane&15; element e<8 -> K = 8*hi + e ; e>=8 -> K = 16 + 8*hi + (e-8)
__device__ __forceinline__ v16h ld_frag_b(const f16_t* baseT, int ld, int kbase) {
  int lane = threadIdx.x & 31;
  int n = lane & 15, hi = lane >> 4;
  const f16_t* p = baseT + n * ld + kbase;
  union { v16h v; v8h h[2]; } u;
  u.h[0] = *(const v8h*)(p + hi * 8);
  u.h[1] = *(const v8h*)(p + 16 + hi * 8);
  return u.v;
}

// ---------------------------------------------------------------------------
// Kernel 1: QKV = x @ W_qkv + b_qkv  (f32 inputs, f16 compute, f32 acc)
// Writes Q, K as [B][H][N][DK] f16 and V transposed as [B][H][DK][N] f16.
// Block tile 128x128, BK=32, 256 threads = 8 waves (4x2 wave grid).
// (Staging stays on the VALU path: it must convert fp32 -> f16.)
// ---------------------------------------------------------------------------
__global__ __launch_bounds__(256) void gemm_qkv_kernel(
    const float* __restrict__ x, const float* __restrict__ Wqkv,
    const float* __restrict__ bqkv,
    f16_t* __restrict__ Qg, f16_t* __restrict__ Kg, f16_t* __restrict__ Vt) {
  constexpr int Kdim = DMODEL;        // 1024
  constexpr int Ncols = 3 * DMODEL;   // 3072
  constexpr int LD = 40;              // 32 + 8 pad (keeps 16B alignment)
  __shared__ alignas(16) f16_t As[128 * LD];
  __shared__ alignas(16) f16_t Bs[128 * LD];

  const int t = threadIdx.x;
  const int lane = t & 31, wv = t >> 5;
  const int wm = wv & 3, wn = wv >> 2;
  const int bn0 = blockIdx.x * 128, bm0 = blockIdx.y * 128;
  const int col = lane & 15, hi = lane >> 4;

  v8f c[2][4];
  for (int j = 0; j < 4; ++j) {
    float bv = bqkv[bn0 + wn * 64 + j * 16 + col];
    for (int i = 0; i < 2; ++i)
      for (int r = 0; r < 8; ++r) c[i][j][r] = bv;
  }

  for (int kt = 0; kt < Kdim / 32; ++kt) {
    const int k0 = kt * 32;
    // Stage A (fp32 -> f16): 128x32, 512 chunks of 8 halves
    for (int i2 = 0; i2 < 2; ++i2) {
      int chunk = i2 * 256 + t;
      int row = chunk >> 2, c8 = chunk & 3;
      const float4* gp =
          (const float4*)(x + (size_t)(bm0 + row) * Kdim + k0 + c8 * 8);
      float4 f0 = gp[0], f1 = gp[1];
      union { v8h v; f16_t e[8]; } u;
      u.e[0] = (f16_t)f0.x; u.e[1] = (f16_t)f0.y;
      u.e[2] = (f16_t)f0.z; u.e[3] = (f16_t)f0.w;
      u.e[4] = (f16_t)f1.x; u.e[5] = (f16_t)f1.y;
      u.e[6] = (f16_t)f1.z; u.e[7] = (f16_t)f1.w;
      *(v8h*)&As[row * LD + c8 * 8] = u.v;
    }
    // Stage B transposed: Bs[n][kk] = W[k0+kk][bn0+n]
    for (int i2 = 0; i2 < 16; ++i2) {
      int id = i2 * 256 + t;
      int kk = id >> 7, n = id & 127;
      Bs[n * LD + kk] = (f16_t)Wqkv[(size_t)(k0 + kk) * Ncols + bn0 + n];
    }
    __syncthreads();

    v16h a0 = ld_frag_a(&As[(wm * 32 + 0) * LD], LD, 0);
    v16h a1 = ld_frag_a(&As[(wm * 32 + 16) * LD], LD, 0);
#pragma unroll
    for (int j = 0; j < 4; ++j) {
      v16h bf = ld_frag_b(&Bs[(wn * 64 + j * 16) * LD], LD, 0);
      c[0][j] = wmma_f16(a0, bf, c[0][j]);
      c[1][j] = wmma_f16(a1, bf, c[1][j]);
    }
    __syncthreads();
  }

  // Epilogue: route columns to Q / K / V(transposed)
  for (int i = 0; i < 2; ++i) {
    int gmBase = bm0 + wm * 32 + i * 16 + hi * 8;  // flat row in [B*N)
    int b = gmBase >> 11;
    int tok = gmBase & 2047;
    for (int j = 0; j < 4; ++j) {
      int gn = bn0 + wn * 64 + j * 16 + col;  // uniform segment per tile
      if (gn < DMODEL) {
        int h = gn >> 6, d = gn & 63;
        f16_t* p = Qg + (((size_t)(b * NHEAD + h) * NSEQ + tok) * DK + d);
        for (int r = 0; r < 8; ++r) p[(size_t)r * DK] = (f16_t)c[i][j][r];
      } else if (gn < 2 * DMODEL) {
        int h = (gn - DMODEL) >> 6, d = gn & 63;
        f16_t* p = Kg + (((size_t)(b * NHEAD + h) * NSEQ + tok) * DK + d);
        for (int r = 0; r < 8; ++r) p[(size_t)r * DK] = (f16_t)c[i][j][r];
      } else {
        int dc = gn - 2 * DMODEL;
        int h = dc >> 6, dd = dc & 63;
        union { v8h v; f16_t e[8]; } u;
        for (int r = 0; r < 8; ++r) u.e[r] = (f16_t)c[i][j][r];
        *(v8h*)(Vt + ((size_t)(b * NHEAD + h) * DK + dd) * NSEQ + tok) = u.v;
      }
    }
  }
}

// ---------------------------------------------------------------------------
// Kernel 2: flash attention. One block = (b, h, 64 q rows); 4 waves,
// each wave owns 16 q rows. Online softmax with row stats in registers.
// K/V tiles double-buffered via async global->LDS copies (ASYNCcnt).
// ---------------------------------------------------------------------------
__global__ __launch_bounds__(128) void attn_kernel(
    const f16_t* __restrict__ Qg, const f16_t* __restrict__ Kg,
    const f16_t* __restrict__ Vt, const int* __restrict__ amask,
    f16_t* __restrict__ Oh) {
  constexpr int LDT = 72;  // 64 + 8 pad
  __shared__ alignas(16) f16_t Qs[64 * LDT];
  __shared__ alignas(16) f16_t Ks[2][64 * LDT];  // [kv][d]  (B^T for Q*K^T)
  __shared__ alignas(16) f16_t Vs[2][64 * LDT];  // [d][kv]  (B^T for P*V)
  __shared__ alignas(16) f16_t Ps[64 * LDT];     // per-wave 16-row strips

  const int bid = blockIdx.x;
  const int qb = bid & 31;
  const int h = (bid >> 5) & 15;
  const int b = bid >> 9;
  const int t = threadIdx.x, lane = t & 31, wv = t >> 5;
  const int col = lane & 15, hi = lane >> 4;

  const f16_t* Qbase = Qg + ((size_t)(b * NHEAD + h) * NSEQ + qb * 64) * DK;
  const f16_t* Kbase = Kg + ((size_t)(b * NHEAD + h) * NSEQ) * DK;
  const f16_t* Vbase = Vt + ((size_t)(b * NHEAD + h) * DK) * NSEQ;
  const int* mk = amask + b * NSEQ;

  auto stage_kv = [&](int buf, int kv0) {
#pragma unroll
    for (int i = 0; i < 4; ++i) {
      int chunk = i * 128 + t;
      int row = chunk >> 3, c8 = chunk & 7;
      cp16_async(&Ks[buf][row * LDT + c8 * 8],
                 Kbase + (size_t)(kv0 + row) * DK + c8 * 8);
      cp16_async(&Vs[buf][row * LDT + c8 * 8],
                 Vbase + (size_t)row * NSEQ + kv0 + c8 * 8);
    }
  };

  // Stage Q (once) and the first K/V tile, all async.
#pragma unroll
  for (int i = 0; i < 4; ++i) {
    int chunk = i * 128 + t;
    int row = chunk >> 3, c8 = chunk & 7;
    cp16_async(&Qs[row * LDT + c8 * 8], Qbase + row * DK + c8 * 8);
  }
  stage_kv(0, 0);
  async_wait0();
  __syncthreads();

  const v16h qa0 = ld_frag_a(&Qs[(wv * 16) * LDT], LDT, 0);
  const v16h qa1 = ld_frag_a(&Qs[(wv * 16) * LDT], LDT, 32);

  float mrow[8], lrow[8];
  v8f o[4];
  for (int r = 0; r < 8; ++r) { mrow[r] = -1e30f; lrow[r] = 0.f; }
  for (int td = 0; td < 4; ++td)
    for (int r = 0; r < 8; ++r) o[td][r] = 0.f;

  for (int j = 0; j < NSEQ / 64; ++j) {
    const int kv0 = j * 64;
    const int cur = j & 1;
    // Prefetch next tile into the other buffer while we compute.
    if (j + 1 < NSEQ / 64) stage_kv(cur ^ 1, kv0 + 64);

    // S strip 16(q) x 64(kv) for this wave
    v8f s[4];
#pragma unroll
    for (int tn = 0; tn < 4; ++tn) {
      v8f z;
      for (int r = 0; r < 8; ++r) z[r] = 0.f;
      v16h kb0 = ld_frag_b(&Ks[cur][(tn * 16) * LDT], LDT, 0);
      v16h kb1 = ld_frag_b(&Ks[cur][(tn * 16) * LDT], LDT, 32);
      z = wmma_f16(qa0, kb0, z);
      z = wmma_f16(qa1, kb1, z);
      int mv = mk[kv0 + tn * 16 + col];  // column fixed per lane
      for (int r = 0; r < 8; ++r)
        s[tn][r] = mv ? z[r] * 0.125f : -1e30f;  // DK^-0.5 = 0.125
    }

    // Online softmax; row r+8*hi lives at element r across a 16-lane group
    float escale[8];
#pragma unroll
    for (int r = 0; r < 8; ++r) {
      float mx = s[0][r];
      for (int tn = 1; tn < 4; ++tn) mx = fmaxf(mx, s[tn][r]);
      for (int d = 1; d < 16; d <<= 1) mx = fmaxf(mx, __shfl_xor(mx, d, 32));
      float nm = fmaxf(mrow[r], mx);
      escale[r] = __expf(mrow[r] - nm);
      mrow[r] = nm;
      float rs = 0.f;
      for (int tn = 0; tn < 4; ++tn) {
        float p = __expf(s[tn][r] - nm);
        s[tn][r] = p;
        rs += p;
      }
      for (int d = 1; d < 16; d <<= 1) rs += __shfl_xor(rs, d, 32);
      lrow[r] = lrow[r] * escale[r] + rs;
    }
    for (int td = 0; td < 4; ++td)
      for (int r = 0; r < 8; ++r) o[td][r] *= escale[r];

    // P -> LDS (wave-private strip), re-enter as A fragment
    for (int tn = 0; tn < 4; ++tn)
      for (int r = 0; r < 8; ++r)
        Ps[(wv * 16 + hi * 8 + r) * LDT + tn * 16 + col] = (f16_t)s[tn][r];

    v16h pa0 = ld_frag_a(&Ps[(wv * 16) * LDT], LDT, 0);
    v16h pa1 = ld_frag_a(&Ps[(wv * 16) * LDT], LDT, 32);
#pragma unroll
    for (int td = 0; td < 4; ++td) {
      v16h vb0 = ld_frag_b(&Vs[cur][(td * 16) * LDT], LDT, 0);
      v16h vb1 = ld_frag_b(&Vs[cur][(td * 16) * LDT], LDT, 32);
      o[td] = wmma_f16(pa0, vb0, o[td]);
      o[td] = wmma_f16(pa1, vb1, o[td]);
    }

    // Own async copies for the next tile must land before anyone reads them.
    async_wait0();
    __syncthreads();
  }

  // Normalize and store O as f16 [B*N][D]
  for (int r = 0; r < 8; ++r) lrow[r] = 1.0f / lrow[r];
  int gmBase = b * NSEQ + qb * 64 + wv * 16 + hi * 8;
  for (int td = 0; td < 4; ++td) {
    int gd = h * DK + td * 16 + col;
    for (int r = 0; r < 8; ++r)
      Oh[(size_t)(gmBase + r) * DMODEL + gd] = (f16_t)(o[td][r] * lrow[r]);
  }
}

// ---------------------------------------------------------------------------
// Kernel 3: out = Oh @ W_out + b_out (f16 A async-staged, fp32 B converted,
// fp32 out). A tile goes straight to LDS via ASYNCcnt path; B conversion
// (VALU) overlaps with it naturally.
// ---------------------------------------------------------------------------
__global__ __launch_bounds__(256) void gemm_out_kernel(
    const f16_t* __restrict__ Oh, const float* __restrict__ Wout,
    const float* __restrict__ bout, float* __restrict__ out) {
  constexpr int Kdim = DMODEL;   // 1024
  constexpr int Ncols = DMODEL;  // 1024
  constexpr int LD = 40;
  __shared__ alignas(16) f16_t As[128 * LD];
  __shared__ alignas(16) f16_t Bs[128 * LD];

  const int t = threadIdx.x;
  const int lane = t & 31, wv = t >> 5;
  const int wm = wv & 3, wn = wv >> 2;
  const int bn0 = blockIdx.x * 128, bm0 = blockIdx.y * 128;
  const int col = lane & 15, hi = lane >> 4;

  v8f c[2][4];
  for (int j = 0; j < 4; ++j) {
    float bv = bout[bn0 + wn * 64 + j * 16 + col];
    for (int i = 0; i < 2; ++i)
      for (int r = 0; r < 8; ++r) c[i][j][r] = bv;
  }

  for (int kt = 0; kt < Kdim / 32; ++kt) {
    const int k0 = kt * 32;
    // A tile: pure f16 copy -> async direct-to-LDS
    for (int i2 = 0; i2 < 2; ++i2) {
      int chunk = i2 * 256 + t;
      int row = chunk >> 2, c8 = chunk & 3;
      cp16_async(&As[row * LD + c8 * 8],
                 Oh + (size_t)(bm0 + row) * Kdim + k0 + c8 * 8);
    }
    // B tile: fp32 -> f16 with transpose (VALU work overlaps the async A)
    for (int i2 = 0; i2 < 16; ++i2) {
      int id = i2 * 256 + t;
      int kk = id >> 7, n = id & 127;
      Bs[n * LD + kk] = (f16_t)Wout[(size_t)(k0 + kk) * Ncols + bn0 + n];
    }
    async_wait0();
    __syncthreads();

    v16h a0 = ld_frag_a(&As[(wm * 32 + 0) * LD], LD, 0);
    v16h a1 = ld_frag_a(&As[(wm * 32 + 16) * LD], LD, 0);
#pragma unroll
    for (int j = 0; j < 4; ++j) {
      v16h bf = ld_frag_b(&Bs[(wn * 64 + j * 16) * LD], LD, 0);
      c[0][j] = wmma_f16(a0, bf, c[0][j]);
      c[1][j] = wmma_f16(a1, bf, c[1][j]);
    }
    __syncthreads();
  }

  for (int i = 0; i < 2; ++i) {
    int gmBase = bm0 + wm * 32 + i * 16 + hi * 8;
    for (int j = 0; j < 4; ++j) {
      int gn = bn0 + wn * 64 + j * 16 + col;
      float* p = out + (size_t)gmBase * Ncols + gn;
      for (int r = 0; r < 8; ++r) p[(size_t)r * Ncols] = c[i][j][r];
    }
  }
}

// ---------------------------------------------------------------------------
extern "C" void kernel_launch(void* const* d_in, const int* in_sizes, int n_in,
                              void* d_out, int out_size, void* d_ws,
                              size_t ws_size, hipStream_t stream) {
  const float* x = (const float*)d_in[0];
  const int* amask = (const int*)d_in[1];
  const float* Wqkv = (const float*)d_in[2];
  const float* bqkv = (const float*)d_in[3];
  const float* Wout = (const float*)d_in[4];
  const float* bout = (const float*)d_in[5];
  float* out = (float*)d_out;

  char* ws = (char*)d_ws;
  const size_t SZ = (size_t)BDIM * NHEAD * NSEQ * DK * sizeof(f16_t);  // 8 MB
  f16_t* Qg = (f16_t*)(ws + 0 * SZ);
  f16_t* Kg = (f16_t*)(ws + 1 * SZ);
  f16_t* Vt = (f16_t*)(ws + 2 * SZ);
  f16_t* Oh = (f16_t*)(ws + 3 * SZ);

  dim3 g1(3 * DMODEL / 128, BDIM * NSEQ / 128);  // (24, 32)
  gemm_qkv_kernel<<<g1, 256, 0, stream>>>(x, Wqkv, bqkv, Qg, Kg, Vt);

  dim3 g2(BDIM * NHEAD * (NSEQ / 64));  // 1024
  attn_kernel<<<g2, 128, 0, stream>>>(Qg, Kg, Vt, amask, Oh);

  dim3 g3(DMODEL / 128, BDIM * NSEQ / 128);  // (8, 32)
  gemm_out_kernel<<<g3, 256, 0, stream>>>(Oh, Wout, bout, out);
}
